// vir_branch_encode_33981781246234
// MI455X (gfx1250) — compile-verified, matrix operand order checked by reading, not run on hardware
//
#include <hip/hip_runtime.h>
#include <hip/hip_bf16.h>
#include <math.h>

typedef __attribute__((ext_vector_type(16))) _Float16 v16h;
typedef __attribute__((ext_vector_type(8)))  float    v8f;
typedef int v4i_ __attribute__((vector_size(16)));

#define B_IMG 8
#define NE 8
#define TOPK 2
#define TDIMF 512

#if __has_builtin(__builtin_amdgcn_global_load_async_to_lds_b128) && \
    __has_builtin(__builtin_amdgcn_s_wait_asynccnt)
#define HAS_ASYNC_LDS 1
#else
#define HAS_ASYNC_LDS 0
#endif

// k index pattern inside one 32-wide K step for 16-bit A/B fragments (wave32):
// VGPR v holds packed f16 pair (j=0 low, j=1 high); lane half g selects +8.
__device__ __forceinline__ int kpat(int v, int j, int g) {
  return ((v >> 2) << 4) + (g << 3) + ((v & 3) << 1) + j;
}

__device__ __forceinline__ float gelu_tanh(float x) {
  return 0.5f * x * (1.0f + tanhf(0.7978845608028654f * (x + 0.044715f * x * x * x)));
}

// ---------------- direct 3x3 conv (pad=1), optional stride-2 + LeakyReLU ----
__global__ void conv3x3_kernel(const float* __restrict__ in, const float* __restrict__ w,
                               const float* __restrict__ bias, float* __restrict__ out,
                               int Cin, int Cout, int Hin, int stride, int Hout, int leaky) {
  long idx = (long)blockIdx.x * blockDim.x + threadIdx.x;
  long total = (long)B_IMG * Cout * Hout * Hout;
  if (idx >= total) return;
  int ox = (int)(idx % Hout); long t = idx / Hout;
  int oy = (int)(t % Hout); t /= Hout;
  int oc = (int)(t % Cout); int b = (int)(t / Cout);
  float acc = bias[oc];
  for (int ic = 0; ic < Cin; ++ic) {
    const float* ip = in + (long)(b * Cin + ic) * Hin * Hin;
    const float* wp = w + (long)(oc * Cin + ic) * 9;
#pragma unroll
    for (int dy = 0; dy < 3; ++dy) {
      int iy = oy * stride + dy - 1;
      if ((unsigned)iy >= (unsigned)Hin) continue;
#pragma unroll
      for (int dx = 0; dx < 3; ++dx) {
        int ix = ox * stride + dx - 1;
        if ((unsigned)ix >= (unsigned)Hin) continue;
        acc = fmaf(ip[(long)iy * Hin + ix], wp[dy * 3 + dx], acc);
      }
    }
  }
  if (leaky) acc = (acc >= 0.f) ? acc : 0.01f * acc;
  out[idx] = acc;
}

// ---------------- global average pool over H*W, one block per (b,c) ---------
__global__ void pool_kernel(const float* __restrict__ in, float* __restrict__ pooled, int HW) {
  int bc = blockIdx.x;
  const float* p = in + (long)bc * HW;
  float s = 0.f;
  for (int i = threadIdx.x; i < HW; i += blockDim.x) s += p[i];
  __shared__ float red[256];
  red[threadIdx.x] = s;
  __syncthreads();
  for (int off = 128; off > 0; off >>= 1) {
    if (threadIdx.x < off) red[threadIdx.x] += red[threadIdx.x + off];
    __syncthreads();
  }
  if (threadIdx.x == 0) pooled[bc] = red[0] / (float)HW;
}

// ---------------- gate: logits -> softmax -> top-2 (one wave per image) -----
__global__ void gate_kernel(const float* __restrict__ pooled, const float* __restrict__ text,
                            const float* __restrict__ gw, const float* __restrict__ gb,
                            const float* __restrict__ te, const int* __restrict__ task_id,
                            int dim, float* __restrict__ probs, float* __restrict__ wts,
                            int* __restrict__ eidx) {
  int b = blockIdx.x;
  int lane = threadIdx.x;
  int rows = dim + TDIMF;
  float acc[NE];
#pragma unroll
  for (int e = 0; e < NE; ++e) acc[e] = 0.f;
  for (int r = lane; r < rows; r += 32) {
    float v = (r < dim) ? pooled[b * dim + r] : text[b * TDIMF + (r - dim)];
    const float* grow = gw + (long)r * NE;
#pragma unroll
    for (int e = 0; e < NE; ++e) acc[e] = fmaf(v, grow[e], acc[e]);
  }
#pragma unroll
  for (int e = 0; e < NE; ++e)
    for (int off = 16; off > 0; off >>= 1) acc[e] += __shfl_xor(acc[e], off, 32);
  if (lane == 0) {
    int task = task_id[0];
    float lg[NE], mx = -1e30f;
    for (int e = 0; e < NE; ++e) { lg[e] = acc[e] + gb[e] + te[task * NE + e]; mx = fmaxf(mx, lg[e]); }
    float sum = 0.f;
    for (int e = 0; e < NE; ++e) { lg[e] = __expf(lg[e] - mx); sum += lg[e]; }
    float p[NE];
    for (int e = 0; e < NE; ++e) { p[e] = lg[e] / sum; probs[b * NE + e] = p[e]; }
    int i0 = 0;
    for (int e = 1; e < NE; ++e) if (p[e] > p[i0]) i0 = e;
    int i1 = (i0 == 0) ? 1 : 0;
    for (int e = 0; e < NE; ++e) if (e != i0 && p[e] > p[i1]) i1 = e;
    float s2 = p[i0] + p[i1];
    wts[b * 2 + 0] = p[i0] / s2;
    wts[b * 2 + 1] = p[i1] / s2;
    eidx[b * 2 + 0] = i0;
    eidx[b * 2 + 1] = i1;
  }
}

// ---------------- MoE pointwise FF via WMMA f32<-f16, residual add ----------
// Block = 128 threads = 4 waves, each wave owns a 16-pixel group of a shared
// 64-pixel tile. The [DIM][64] f32 x-tile is staged into LDS once, coalesced,
// through the CDNA5 async-to-LDS path (ASYNCcnt) when available; fragments,
// the residual read, and the h restripe all come from LDS afterwards.
template <int DIM, int HID>
__global__ void __launch_bounds__(128)
moe_kernel(const float* __restrict__ in, float* __restrict__ out,
           const float* __restrict__ w1, const float* __restrict__ b1,
           const float* __restrict__ w2, const float* __restrict__ b2,
           const int* __restrict__ eidx, const float* __restrict__ wts, int HW) {
  constexpr int KS1 = (DIM + 31) / 32;   // K steps of first matmul (zero-padded)
  constexpr int KS2 = HID / 32;          // K steps of second matmul
  constexpr int MT1 = HID / 16;          // M tiles of first matmul
  constexpr int MT2 = DIM / 16;          // M tiles of second matmul
  constexpr int PADW = 68;               // 64 px + 4 pad: bank = (4c+px)%64, row 272B (16B-mult)
  __shared__ __align__(16) float ldsX[DIM * PADW];
  __shared__ _Float16 ldsH[4 * HID * 16];

  int tilesPerImg = HW / 64;
  int b = blockIdx.x / tilesPerImg;
  int tile = blockIdx.x % tilesPerImg;
  int wave = threadIdx.x >> 5;
  int lane = threadIdx.x & 31;
  int g = lane >> 4;
  int n = lane & 15;
  int pxl = wave * 16 + n;               // pixel within the 64-px tile
  int pix = tile * 64 + pxl;             // pixel within the image
  _Float16* hslab = &ldsH[wave * HID * 16];

  // ---- stage x tile [DIM][64] f32 into LDS (16B granules, coalesced) ----
  const float* gbase = in + (long)b * DIM * HW + (long)tile * 64;
  {
    constexpr int NVEC = DIM * 16;       // b128 granules: 16 per channel row
    for (int vi = threadIdx.x; vi < NVEC; vi += 128) {
      int c = vi >> 4, q = vi & 15;
      const float* gp = gbase + (long)c * HW + q * 4;
      float* lp = &ldsX[c * PADW + q * 4];
#if HAS_ASYNC_LDS
      __builtin_amdgcn_global_load_async_to_lds_b128(
          (__attribute__((address_space(1))) v4i_*)(void*)gp,
          (__attribute__((address_space(3))) v4i_*)(void*)lp, 0, 0);
#else
      float4 d = *(const float4*)gp;
      lp[0] = d.x; lp[1] = d.y; lp[2] = d.z; lp[3] = d.w;
#endif
    }
#if HAS_ASYNC_LDS
    __builtin_amdgcn_s_wait_asynccnt(0);
#endif
  }
  __syncthreads();

  // activation B-fragments for first matmul (K = input channels), from LDS
  v16h xfrag[KS1];
#pragma unroll
  for (int ks = 0; ks < KS1; ++ks)
#pragma unroll
    for (int v = 0; v < 8; ++v)
#pragma unroll
      for (int j = 0; j < 2; ++j) {
        int c = ks * 32 + kpat(v, j, g);
        float val = (c < DIM) ? ldsX[c * PADW + pxl] : 0.f;
        xfrag[ks][v * 2 + j] = (_Float16)val;
      }

  v8f oacc[MT2];
#pragma unroll
  for (int m = 0; m < MT2; ++m)
#pragma unroll
    for (int r = 0; r < 8; ++r) oacc[m][r] = 0.f;

  for (int kk = 0; kk < TOPK; ++kk) {
    int e = eidx[b * 2 + kk];
    float wt = wts[b * 2 + kk];
    // --- h = gelu(W1 x + b1) -> LDS slab (f16)
#pragma unroll
    for (int mt = 0; mt < MT1; ++mt) {
      v8f acc;
#pragma unroll
      for (int r = 0; r < 8; ++r) acc[r] = 0.f;
      int row = mt * 16 + n;
      const float* wrow = w1 + ((long)e * HID + row) * DIM;
#pragma unroll
      for (int ks = 0; ks < KS1; ++ks) {
        v16h a;
#pragma unroll
        for (int v = 0; v < 8; ++v)
#pragma unroll
          for (int j = 0; j < 2; ++j) {
            int k = ks * 32 + kpat(v, j, g);
            a[v * 2 + j] = (_Float16)((k < DIM) ? wrow[k] : 0.f);
          }
        acc = __builtin_amdgcn_wmma_f32_16x16x32_f16(false, a, false, xfrag[ks],
                                                     (short)0, acc, false, false);
      }
#pragma unroll
      for (int r = 0; r < 8; ++r) {
        int m = mt * 16 + r + 8 * g;            // D layout: M = r + 8*g
        float hv = gelu_tanh(acc[r] + b1[e * HID + m]);
        hslab[m * 16 + n] = (_Float16)hv;
      }
    }
    __syncthreads();
    // --- restripe h into B-fragments (K = hid)
    v16h hfrag[KS2];
#pragma unroll
    for (int ks = 0; ks < KS2; ++ks)
#pragma unroll
      for (int v = 0; v < 8; ++v)
#pragma unroll
        for (int j = 0; j < 2; ++j) {
          int k = ks * 32 + kpat(v, j, g);
          hfrag[ks][v * 2 + j] = hslab[k * 16 + n];
        }
    // --- y = W2 h + b2 ; accumulate wt*y
#pragma unroll
    for (int mt = 0; mt < MT2; ++mt) {
      v8f acc;
#pragma unroll
      for (int r = 0; r < 8; ++r) acc[r] = 0.f;
      int row = mt * 16 + n;
      const float* wrow = w2 + ((long)e * DIM + row) * HID;
#pragma unroll
      for (int ks = 0; ks < KS2; ++ks) {
        v16h a;
#pragma unroll
        for (int v = 0; v < 8; ++v)
#pragma unroll
          for (int j = 0; j < 2; ++j) {
            int k = ks * 32 + kpat(v, j, g);
            a[v * 2 + j] = (_Float16)wrow[k];
          }
        acc = __builtin_amdgcn_wmma_f32_16x16x32_f16(false, a, false, hfrag[ks],
                                                     (short)0, acc, false, false);
      }
#pragma unroll
      for (int r = 0; r < 8; ++r) {
        int c = mt * 16 + r + 8 * g;
        oacc[mt][r] = fmaf(wt, acc[r] + b2[e * DIM + c], oacc[mt][r]);
      }
    }
    __syncthreads();
  }
  // --- residual write-out (residual x read from LDS, not HBM)
#pragma unroll
  for (int mt = 0; mt < MT2; ++mt)
#pragma unroll
    for (int r = 0; r < 8; ++r) {
      int c = mt * 16 + r + 8 * g;
      out[(long)(b * DIM + c) * HW + pix] = ldsX[c * PADW + pxl] + oacc[mt][r];
    }
}

// ---------------- mi loss accumulation over all 4 stages --------------------
__global__ void mi_kernel(const float* __restrict__ probs, float* __restrict__ out_mi) {
  if (threadIdx.x == 0 && blockIdx.x == 0) {
    float total = 0.f;
    for (int s = 0; s < 4; ++s) {
      float acc = 0.f;
      for (int e = 0; e < NE; ++e) {
        float m = 0.f;
        for (int b = 0; b < B_IMG; ++b) m += probs[(s * B_IMG + b) * NE + e];
        m /= (float)B_IMG;
        acc += m * m;
      }
      total += (float)NE * acc;
    }
    *out_mi = total;
  }
}

extern "C" void kernel_launch(void* const* d_in, const int* in_sizes, int n_in,
                              void* d_out_v, int out_size, void* d_ws, size_t ws_size,
                              hipStream_t stream) {
  (void)in_sizes; (void)n_in; (void)out_size; (void)ws_size;
  const float* x      = (const float*)d_in[0];
  const float* text   = (const float*)d_in[1];
  const int*   task   = (const int*)d_in[2];
  const float* conv_w = (const float*)d_in[3];
  const float* conv_b = (const float*)d_in[4];
  float* d_out = (float*)d_out_v;

  const int dims[4] = {16, 32, 48, 64};
  const int Hs[4]   = {256, 128, 64, 32};
  const long out_off[4] = {0, 4194304, 5767168, 6291456};
  const long mi_off = 6455296;  // 6291456 + 8*80*16*16

  char* ws = (char*)d_ws;
  float* bufA   = (float*)ws;                    // stem output, 8*16*256*256 f32
  float* bufB   = (float*)(ws + 33554432);       // MoE output ping buffer
  float* pooled = (float*)(ws + 67108864);       // up to 8*64
  float* probs  = pooled + 1024;                 // 4*8*8
  float* wtsb   = probs + 256;                   // 4*16
  int*   idxb   = (int*)(wtsb + 64);             // 4*16

  // stem conv: 1 -> 16, stride 1 (no activation)
  {
    long total = (long)B_IMG * 16 * 256 * 256;
    conv3x3_kernel<<<(int)((total + 255) / 256), 256, 0, stream>>>(
        x, conv_w, conv_b, bufA, 1, 16, 256, 1, 256, 0);
  }

  const float* stage_in = bufA;
  for (int s = 0; s < 4; ++s) {
    int dim = dims[s], outc = dim + 16;
    int H = Hs[s], HW = H * H, Hout = H / 2;
    const float* gw = (const float*)d_in[5 + s * 9 + 0];
    const float* gb = (const float*)d_in[5 + s * 9 + 1];
    const float* te = (const float*)d_in[5 + s * 9 + 2];
    const float* w1 = (const float*)d_in[5 + s * 9 + 3];
    const float* b1 = (const float*)d_in[5 + s * 9 + 4];
    const float* w2 = (const float*)d_in[5 + s * 9 + 5];
    const float* b2 = (const float*)d_in[5 + s * 9 + 6];
    const float* dw = (const float*)d_in[5 + s * 9 + 7];
    const float* db = (const float*)d_in[5 + s * 9 + 8];

    pool_kernel<<<B_IMG * dim, 256, 0, stream>>>(stage_in, pooled, HW);
    gate_kernel<<<B_IMG, 32, 0, stream>>>(pooled, text, gw, gb, te, task, dim,
                                          probs + s * B_IMG * NE, wtsb + s * 16, idxb + s * 16);

    int moe_blocks = B_IMG * (HW / 64);
    switch (s) {
      case 0: moe_kernel<16, 32><<<moe_blocks, 128, 0, stream>>>(
                  stage_in, bufB, w1, b1, w2, b2, idxb + s * 16, wtsb + s * 16, HW); break;
      case 1: moe_kernel<32, 64><<<moe_blocks, 128, 0, stream>>>(
                  stage_in, bufB, w1, b1, w2, b2, idxb + s * 16, wtsb + s * 16, HW); break;
      case 2: moe_kernel<48, 96><<<moe_blocks, 128, 0, stream>>>(
                  stage_in, bufB, w1, b1, w2, b2, idxb + s * 16, wtsb + s * 16, HW); break;
      case 3: moe_kernel<64, 128><<<moe_blocks, 128, 0, stream>>>(
                  stage_in, bufB, w1, b1, w2, b2, idxb + s * 16, wtsb + s * 16, HW); break;
    }

    long ctotal = (long)B_IMG * outc * Hout * Hout;
    conv3x3_kernel<<<(int)((ctotal + 255) / 256), 256, 0, stream>>>(
        bufB, dw, db, d_out + out_off[s], dim, outc, H, 2, Hout, 1);
    stage_in = d_out + out_off[s];
  }
  mi_kernel<<<1, 32, 0, stream>>>(probs, d_out + mi_off);
}
